// LSTMLayer_9801115369773
// MI455X (gfx1250) — compile-verified, matrix operand order checked by reading, not run on hardware
//
#include <hip/hip_runtime.h>

// LSTM layer for MI455X (gfx1250, wave32).
// Phase 1: x_gates = X @ W_ih^T + (b_ih + b_hh)  -- bf16 WMMA GEMM with
//          Tensor-Data-Mover (tensor_load_to_lds) tile staging.
// Phase 2: persistent 16-workgroup cluster kernel; W_hh slice resident in LDS,
//          h staged per-step into LDS via TDM, per-step 32x2048 gate GEMM via
//          v_wmma_f32_16x16x32_bf16, cluster barriers (s_barrier -3) between
//          GEMM and elementwise gate/state update.
//
// Workspace layout (requires ws_size >= ~581 MB):
//   [0, 512MB)   x_gates fp32 [T,B,4H]
//   +512MB:      X bf16 (64MB), W_ih bf16 (2MB), W_hh bf16 (2MB),
//                h bf16 (32KB), c fp32 (64KB), gate scratch fp32 (256KB)

#define T_STEPS 2048
#define BATCH   32
#define HID     512
#define GATES   2048  // 4*HID

typedef __attribute__((ext_vector_type(16))) __bf16          bf16x16;
typedef __attribute__((ext_vector_type(8)))  float           f32x8;
typedef __attribute__((ext_vector_type(4)))  unsigned int    u32x4;
typedef __attribute__((ext_vector_type(8)))  unsigned int    u32x8;

union FragBF { u32x4 q[2]; bf16x16 v; };

__device__ __forceinline__ unsigned short f2bf(float x) {
    union { float f; unsigned int u; } v; v.f = x;
    unsigned int r = v.u + 0x7FFFu + ((v.u >> 16) & 1u);  // RNE
    return (unsigned short)(r >> 16);
}

__device__ __forceinline__ float sigmoidf(float x) {
    return 1.0f / (1.0f + __expf(-x));
}

// ---------------------------------------------------------------------------
// Tensor Data Mover: 2-D bf16 tile load Global -> LDS (ISA ch.8 / §15 TDM).
// D# group0 (4 SGPRs): count=1 | lds_addr | global_addr[56:0] | type=2.
// D# group1 (8 SGPRs): data_size=2B, tensor_dim0/1, tile_dim0/1, dim0 stride.
// Issued by one wave; completion tracked with TENSORcnt.
// ---------------------------------------------------------------------------
__device__ __forceinline__ void tdm_load_2d_bf16(
    unsigned lds_off, const void* gaddr,
    unsigned tensor_d0, unsigned tensor_d1,
    unsigned tile_d0, unsigned tile_d1, unsigned stride_d0)
{
    unsigned long long ga = (unsigned long long)gaddr;
    u32x4 g0;
    g0[0] = 1u;                                                 // count=1 (valid), user mode
    g0[1] = lds_off;                                            // lds_addr (bytes)
    g0[2] = (unsigned)ga;                                       // global_addr[31:0]
    g0[3] = ((unsigned)(ga >> 32) & 0x01FFFFFFu) | 0x80000000u; // addr[56:32] | type=2
    u32x8 g1;
    g1[0] = 1u << 16;                                           // data_size=1 -> 2 bytes
    g1[1] = (tensor_d0 & 0xFFFFu) << 16;                        // tensor_dim0[15:0]
    g1[2] = (tensor_d0 >> 16) | ((tensor_d1 & 0xFFFFu) << 16);  // td0[31:16] | td1[15:0]
    g1[3] = (tensor_d1 >> 16) | (tile_d0 << 16);                // td1[31:16] | tile_dim0
    g1[4] = tile_d1 & 0xFFFFu;                                  // tile_dim1 (tile_dim2=0)
    g1[5] = stride_d0;                                          // tensor_dim0_stride[31:0]
    g1[6] = 0u;
    g1[7] = 0u;                                                 // tensor_dim1_stride=0 (2D)
    asm volatile("tensor_load_to_lds %0, %1" :: "s"(g0), "s"(g1) : "memory");
}

// ---------------------------------------------------------------------------
// Kernel 1: fp32 -> bf16 conversion of X / W_ih / W_hh + state init.
// ---------------------------------------------------------------------------
__global__ __launch_bounds__(256) void prep_kernel(
    const float* __restrict__ X, const float* __restrict__ W_ih,
    const float* __restrict__ W_hh, const float* __restrict__ h0,
    const float* __restrict__ c0, unsigned short* __restrict__ x_bf,
    unsigned short* __restrict__ wih_bf, unsigned short* __restrict__ whh_bf,
    unsigned short* __restrict__ h_bf, float* __restrict__ cbuf)
{
    const size_t idx    = (size_t)blockIdx.x * 256 + threadIdx.x;
    const size_t stride = (size_t)gridDim.x * 256;
    for (size_t i = idx; i < (size_t)T_STEPS * BATCH * HID; i += stride)
        x_bf[i] = f2bf(X[i]);
    if (idx < (size_t)GATES * HID) {          // 1048576 elements each
        wih_bf[idx] = f2bf(W_ih[idx]);
        whh_bf[idx] = f2bf(W_hh[idx]);
    }
    if (idx < BATCH * HID) {
        h_bf[idx] = f2bf(h0[idx]);
        cbuf[idx] = c0[idx];
    }
}

// ---------------------------------------------------------------------------
// Kernel 2: x_gates[r, n] = sum_k X[r,k] * W_ih[n,k] + b_ih[n] + b_hh[n]
//   128x128 block tile, K-chunks of 32. A/B tiles staged by TDM (wave 0
//   issues tensor_load_to_lds; all waves consume after s_wait_tensorcnt +
//   workgroup barrier). 8 waves; wave w -> rows [16w,16w+16) x 128 cols.
//   All 8 B fragments are loaded into distinct registers before the WMMA
//   chain so the 8 matrix ops issue back-to-back behind one s_wait_dscnt.
// ---------------------------------------------------------------------------
__global__ __launch_bounds__(256) void xgates_gemm_kernel(
    const unsigned short* __restrict__ Xb, const unsigned short* __restrict__ Wb,
    const float* __restrict__ b_ih, const float* __restrict__ b_hh,
    float* __restrict__ xg)
{
    __shared__ unsigned short Asub[128 * 32];  // [m][k], bf16, 8 KB
    __shared__ unsigned short Bsub[128 * 32];  // [n][k], bf16, 8 KB

    const int tid    = threadIdx.x;
    const int wave   = tid >> 5;
    const int lane   = tid & 31;
    const int laneHi = (lane >> 4) & 1;
    const int rbase  = blockIdx.y * 128;
    const int nbase  = blockIdx.x * 128;

    const unsigned a_lds = (unsigned)(size_t)(void*)Asub;
    const unsigned b_lds = (unsigned)(size_t)(void*)Bsub;

    f32x8 acc[8];
    #pragma unroll
    for (int i = 0; i < 8; ++i) { f32x8 z = {}; acc[i] = z; }

    for (int kc = 0; kc < HID / 32; ++kc) {
        const int kbase = kc * 32;
        if (wave == 0) {
            // 128 rows x 32 cols bf16 tiles; row stride HID elements.
            tdm_load_2d_bf16(a_lds, Xb + (size_t)rbase * HID + kbase,
                             HID, 128, 32, 128, HID);
            tdm_load_2d_bf16(b_lds, Wb + (size_t)nbase * HID + kbase,
                             HID, 128, 32, 128, HID);
            __builtin_amdgcn_s_wait_tensorcnt(0);
        }
        __syncthreads();

        // A fragment: 16x32 bf16; lane<16 holds K{0..7,16..23}, lane>=16 K{8..15,24..31}
        FragBF af;
        {
            const int am = wave * 16 + (lane & 15);
            const int ao = am * 32 + laneHi * 8;
            af.q[0] = *reinterpret_cast<const u32x4*>(&Asub[ao]);
            af.q[1] = *reinterpret_cast<const u32x4*>(&Asub[ao + 16]);
        }
        // Load ALL B fragments first (distinct registers), then WMMA chain.
        FragBF bfr[8];
        #pragma unroll
        for (int nt = 0; nt < 8; ++nt) {
            const int bn = nt * 16 + (lane & 15);
            const int bo = bn * 32 + laneHi * 8;
            bfr[nt].q[0] = *reinterpret_cast<const u32x4*>(&Bsub[bo]);
            bfr[nt].q[1] = *reinterpret_cast<const u32x4*>(&Bsub[bo + 16]);
        }
        #pragma unroll
        for (int nt = 0; nt < 8; ++nt) {
            acc[nt] = __builtin_amdgcn_wmma_f32_16x16x32_bf16(
                false, af.v, false, bfr[nt].v, (short)0, acc[nt], false, false);
        }
        __syncthreads();
    }

    // Epilogue: C/D layout -> VGPR v holds M = v (+8 for lanes 16..31), N = lane&15
    #pragma unroll
    for (int nt = 0; nt < 8; ++nt) {
        const int col  = nbase + nt * 16 + (lane & 15);
        const float bs = b_ih[col] + b_hh[col];
        #pragma unroll
        for (int v = 0; v < 8; ++v) {
            const int row = rbase + wave * 16 + v + laneHi * 8;
            xg[(size_t)row * GATES + col] = acc[nt][v] + bs;
        }
    }
}

// ---------------------------------------------------------------------------
// Kernel 3: persistent recurrence. 16 workgroups = one workgroup-cluster.
//   Workgroup g owns gate columns [g*128, g*128+128): W_hh slice (128x512 bf16
//   = 128 KB) lives in dynamic LDS for all 2048 steps; h (32x512 bf16 = 32 KB)
//   is TDM-staged into static LDS each step. Per step:
//     GEMM  : gates_hw[b, g*128+n] = sum_k h[b,k] * Whh[g*128+n, k]  (WMMA)
//     sync  : device fence + cluster barrier (s_barrier_signal/wait -3)
//     update: wg g applies i/f/g/o for h-columns [g*32, g*32+32), updates
//             c (fp32 ws) and h (fp32 -> d_out, bf16 -> ws), sync again.
//   NOTE: assumes the 16-workgroup grid is dispatched as one cluster.
// ---------------------------------------------------------------------------
__global__ __launch_bounds__(256) void lstm_scan_kernel(
    const float* __restrict__ xg, const unsigned short* __restrict__ Whh_bf,
    unsigned short* __restrict__ h_bf, float* __restrict__ cbuf,
    float* __restrict__ scratch, float* __restrict__ out)
{
    extern __shared__ unsigned short wsub[];      // [128][512] bf16 = 128 KB
    __shared__ unsigned short hsub[BATCH * HID];  // 32 KB

    const int wg     = blockIdx.x;             // 0..15
    const int tid    = threadIdx.x;
    const int wave   = tid >> 5;
    const int lane   = tid & 31;
    const int laneHi = (lane >> 4) & 1;

    const unsigned h_lds = (unsigned)(size_t)(void*)hsub;

    // Load W_hh slice: rows [wg*128, wg*128+128) are contiguous (stride 512).
    if (wave == 0) {
        tdm_load_2d_bf16((unsigned)(size_t)(void*)wsub,
                         Whh_bf + (size_t)wg * 128 * HID,
                         HID, 128, HID, 128, HID);   // full rows, linear copy
        __builtin_amdgcn_s_wait_tensorcnt(0);
    }
    __syncthreads();

    // 16 output tiles (2 Mtiles x 8 Ntiles of 16x16) over 8 waves: 2 tiles/wave.
    const int mt   = wave & 1;
    const int ntb  = (wave >> 1) * 2;
    const int arow = mt * 16 + (lane & 15);      // batch row for A fragment
    const int n0   = ntb * 16 + (lane & 15);     // local gate col for B frag 0

    for (int t = 0; t < T_STEPS; ++t) {
        // Stage current h (bf16, 32x512) into LDS via TDM.
        if (wave == 0) {
            tdm_load_2d_bf16(h_lds, h_bf, HID, BATCH, HID, BATCH, HID);
            __builtin_amdgcn_s_wait_tensorcnt(0);
        }
        __syncthreads();

        f32x8 acc0 = {}; f32x8 acc1 = {};
        #pragma unroll 4
        for (int kc = 0; kc < HID / 32; ++kc) {
            const int kbase = kc * 32;
            FragBF af;
            const unsigned short* hp = hsub + arow * HID + kbase + laneHi * 8;
            af.q[0] = *reinterpret_cast<const u32x4*>(hp);
            af.q[1] = *reinterpret_cast<const u32x4*>(hp + 16);
            FragBF b0, b1;
            const unsigned short* wp0 = wsub + n0 * HID + kbase + laneHi * 8;
            b0.q[0] = *reinterpret_cast<const u32x4*>(wp0);
            b0.q[1] = *reinterpret_cast<const u32x4*>(wp0 + 16);
            const unsigned short* wp1 = wp0 + 16 * HID;
            b1.q[0] = *reinterpret_cast<const u32x4*>(wp1);
            b1.q[1] = *reinterpret_cast<const u32x4*>(wp1 + 16);
            acc0 = __builtin_amdgcn_wmma_f32_16x16x32_bf16(
                false, af.v, false, b0.v, (short)0, acc0, false, false);
            acc1 = __builtin_amdgcn_wmma_f32_16x16x32_bf16(
                false, af.v, false, b1.v, (short)0, acc1, false, false);
        }

        // Spill h@Whh^T partial for this wg's 128 columns to L2 scratch.
        #pragma unroll
        for (int v = 0; v < 8; ++v) {
            const int b   = mt * 16 + v + laneHi * 8;
            const int col = wg * 128 + n0;
            scratch[b * GATES + col]      = acc0[v];
            scratch[b * GATES + col + 16] = acc1[v];
        }
        __threadfence();
        __syncthreads();
        __builtin_amdgcn_s_cluster_barrier();

        // Elementwise gate/state update for h-columns [wg*32, wg*32+32).
        const float* xgt = xg + (size_t)t * BATCH * GATES;
        #pragma unroll
        for (int i = 0; i < 4; ++i) {
            const int e = tid + i * 256;       // 0..1023
            const int b = e >> 5;              // 0..31
            const int j = wg * 32 + (e & 31);  // h column
            const float* xr = xgt + (size_t)b * GATES;
            const float* sr = scratch + (size_t)b * GATES;
            const float iv = sigmoidf(xr[j]         + sr[j]);
            const float fv = sigmoidf(xr[HID + j]   + sr[HID + j]);
            const float gv = tanhf   (xr[2*HID + j] + sr[2*HID + j]);
            const float ov = sigmoidf(xr[3*HID + j] + sr[3*HID + j]);
            const float cn = fv * cbuf[b * HID + j] + iv * gv;
            cbuf[b * HID + j] = cn;
            const float hv = ov * tanhf(cn);
            out[(size_t)t * BATCH * HID + b * HID + j] = hv;
            h_bf[b * HID + j] = f2bf(hv);
            if (t == T_STEPS - 1) {
                const size_t tail = (size_t)T_STEPS * BATCH * HID;
                out[tail + b * HID + j]               = hv;  // h_T
                out[tail + BATCH * HID + b * HID + j] = cn;  // c_T
            }
        }
        __threadfence();
        __syncthreads();
        __builtin_amdgcn_s_cluster_barrier();
    }
}

// ---------------------------------------------------------------------------
extern "C" void kernel_launch(void* const* d_in, const int* in_sizes, int n_in,
                              void* d_out, int out_size, void* d_ws, size_t ws_size,
                              hipStream_t stream) {
    const float* input = (const float*)d_in[0];  // [T,B,I]
    const float* h0    = (const float*)d_in[1];  // [B,H]
    const float* c0    = (const float*)d_in[2];  // [B,H]
    const float* W_ih  = (const float*)d_in[3];  // [4H,I]
    const float* W_hh  = (const float*)d_in[4];  // [4H,H]
    const float* b_ih  = (const float*)d_in[5];  // [4H]
    const float* b_hh  = (const float*)d_in[6];  // [4H]
    float* out = (float*)d_out;                  // [T,B,H] ++ [B,H] ++ [B,H]

    const size_t XG_ELEMS = (size_t)T_STEPS * BATCH * GATES;
    const size_t X_ELEMS  = (size_t)T_STEPS * BATCH * HID;
    float*          xg      = (float*)d_ws;
    unsigned short* x_bf    = (unsigned short*)((char*)d_ws + XG_ELEMS * sizeof(float));
    unsigned short* wih_bf  = x_bf + X_ELEMS;
    unsigned short* whh_bf  = wih_bf + (size_t)GATES * HID;
    unsigned short* h_bf    = whh_bf + (size_t)GATES * HID;
    float*          cbuf    = (float*)(h_bf + BATCH * HID);
    float*          scratch = cbuf + BATCH * HID;

    prep_kernel<<<8192, 256, 0, stream>>>(
        input, W_ih, W_hh, h0, c0, x_bf, wih_bf, whh_bf, h_bf, cbuf);

    dim3 g2(GATES / 128, (T_STEPS * BATCH) / 128, 1);
    xgates_gemm_kernel<<<g2, 256, 0, stream>>>(x_bf, wih_bf, b_ih, b_hh, xg);

    lstm_scan_kernel<<<16, 256, 128 * HID * sizeof(unsigned short), stream>>>(
        xg, whh_bf, h_bf, cbuf, scratch, out);
}
